// PPN_24893630448261
// MI455X (gfx1250) — matrix operand order repaired and checked by reading, not compiled, not run on hardware
//
#include <hip/hip_runtime.h>
#include <hip/hip_bf16.h>

typedef __attribute__((ext_vector_type(16))) _Float16 v16h;
typedef __attribute__((ext_vector_type(8)))  float    v8f;

#define NN 100000
#define EE 1600000
#define CC 40
#define DD 64
#define KPAD 64            // C padded 40 -> 64 for two K=32 WMMA steps
#define NLAYERS 3
#define ALPHA_F 0.9f

// ---------------- generic scalar zero ----------------
__global__ void lp_zero_kernel(float* __restrict__ p, int n) {
    int i = blockIdx.x * blockDim.x + threadIdx.x;
    int stride = gridDim.x * blockDim.x;
    for (; i < n; i += stride) p[i] = 0.0f;
}

// ---------------- float4 zero (n4 = count of float4) ----------------
__global__ void lp_zero4_kernel(float4* __restrict__ p, int n4) {
    int i = blockIdx.x * blockDim.x + threadIdx.x;
    int stride = gridDim.x * blockDim.x;
    float4 z = make_float4(0.f, 0.f, 0.f, 0.f);
    for (; i < n4; i += stride) p[i] = z;
}

// ---------------- in-degree via float atomics ----------------
__global__ void lp_degree_kernel(const int* __restrict__ col, float* __restrict__ deg, int e) {
    int i = blockIdx.x * blockDim.x + threadIdx.x;
    int stride = gridDim.x * blockDim.x;
    for (; i < e; i += stride) atomicAdd(&deg[col[i]], 1.0f);
}

// ---------------- deg -> deg^{-1/2} in place ----------------
__global__ void lp_dinv_kernel(float* __restrict__ deg, int n) {
    int i = blockIdx.x * blockDim.x + threadIdx.x;
    int stride = gridDim.x * blockDim.x;
    for (; i < n; i += stride) {
        float d = deg[i];
        deg[i] = (d > 0.0f) ? rsqrtf(d) : 0.0f;
    }
}

// ---------------- per-edge symmetric norm ----------------
__global__ void lp_norm_kernel(const int* __restrict__ row, const int* __restrict__ col,
                               const float* __restrict__ dinv, float* __restrict__ norm, int e) {
    int i = blockIdx.x * blockDim.x + threadIdx.x;
    int stride = gridDim.x * blockDim.x;
    for (; i < e; i += stride) norm[i] = dinv[row[i]] * dinv[col[i]];
}

// ---------------- prototype init as one-hot GEMM via WMMA ----------------
// y0[N,D] = onehot(labels; train_mask)[N,C] @ protos[C,D], C padded 40->64.
// Protos staged once per block into LDS, transposed + zero-padded, f16:
//   sB[n*KPAD + k]  (n = output dim 0..63, k = class 0..63)
// In this layout every B fragment (fixed n, 16 consecutive k) is one aligned
// 32-byte LDS vector load -> branch-free, no predicated global loads.
__global__ void lp_init_wmma_kernel(const int* __restrict__ train_mask,
                                    const float* __restrict__ protos,
                                    const int* __restrict__ labels,
                                    float* __restrict__ out,
                                    float* __restrict__ res) {
    __shared__ __align__(32) _Float16 sB[DD * KPAD];   // 8 KB

    // cooperative fill: contiguous LDS writes; protos (10 KB) is L2-resident
    for (int t = threadIdx.x; t < DD * KPAD; t += blockDim.x) {
        int n = t >> 6;          // output dim
        int k = t & 63;          // class (padded)
        float v = (k < CC) ? protos[k * DD + n] : 0.0f;
        sB[t] = (_Float16)v;
    }
    __syncthreads();

    int wave = (int)((blockIdx.x * blockDim.x + threadIdx.x) >> 5);
    int lane = (int)(threadIdx.x & 31);
    const int ngroups = NN / 16;            // 6250, exact
    if (wave >= ngroups) return;            // wave-uniform: EXEC all-ones for live waves

    const bool hi   = lane >= 16;
    const int  ncol = lane & 15;
    const int  node = wave * 16 + ncol;     // M = lane&15 per A-matrix layout

    int lbl = labels[node];
    int tm  = train_mask[node];
    int t   = tm ? lbl : -1;                // -1 => all-zero one-hot row

    // A fragments (16x32 f16): lanes0-15 K=0..15, lanes16-31 K=16..31 (matches B).
    v16h a0, a1;
#pragma unroll
    for (int h = 0; h < 16; ++h) {
        int k = (hi ? 16 : 0) + h;
        a0[h] = (k == t)        ? (_Float16)1.0f : (_Float16)0.0f;
        a1[h] = ((k + 32) == t) ? (_Float16)1.0f : (_Float16)0.0f;
    }

    const int kofs = hi ? 16 : 0;           // B layout: lanes0-15 K=0..15, lanes16-31 K=16..31

#pragma unroll
    for (int nt = 0; nt < 4; ++nt) {
        const _Float16* brow = sB + (nt * 16 + ncol) * KPAD;   // fixed output dim for this lane
        v8f c = {};
#pragma unroll
        for (int ks = 0; ks < 2; ++ks) {
            // 16 consecutive k-halves -> one aligned 32-byte LDS vector load
            v16h b = *reinterpret_cast<const v16h*>(brow + ks * 32 + kofs);
            c = __builtin_amdgcn_wmma_f32_16x16x32_f16(
                    /*neg_a=*/false, (ks == 0 ? a0 : a1),
                    /*neg_b=*/false, b,
                    /*c_mod=*/(short)0, c,
                    /*reuse_a=*/false, /*reuse_b=*/false);
        }
        // D layout (32-bit C/D 16x16): lanes0-15 N=lane, VGPR r -> M=r; lanes16-31 -> M=r+8
#pragma unroll
        for (int r = 0; r < 8; ++r) {
            int m   = hi ? (r + 8) : r;
            int idx = (wave * 16 + m) * DD + nt * 16 + ncol;
            float v = c[r];
            out[idx] = v;
            res[idx] = (1.0f - ALPHA_F) * v;
        }
    }
}

// ---------------- edge scatter: agg[col] += norm[e] * out[row] ----------------
// 16 consecutive lanes = one edge (coalesced float4 row gather), 4 f32 atomics each.
__global__ void lp_scatter_kernel(const int* __restrict__ row, const int* __restrict__ col,
                                  const float* __restrict__ norm,
                                  const float* __restrict__ out, float* __restrict__ agg) {
    const int total = EE * 16;
    int i = blockIdx.x * blockDim.x + threadIdx.x;
    int stride = gridDim.x * blockDim.x;
    for (; i < total; i += stride) {
        int e  = i >> 4;
        int d0 = (i & 15) * 4;
        float nv = norm[e];
        const float4* src = reinterpret_cast<const float4*>(out + (size_t)row[e] * DD + d0);
        float4 v = *src;
        float* dst = agg + (size_t)col[e] * DD + d0;
        atomicAdd(dst + 0, nv * v.x);
        atomicAdd(dst + 1, nv * v.y);
        atomicAdd(dst + 2, nv * v.z);
        atomicAdd(dst + 3, nv * v.w);
    }
}

// ---------------- out = clamp(alpha*agg + res, 0, 1), float4 ----------------
__global__ void lp_combine_kernel(const float4* __restrict__ agg, const float4* __restrict__ res,
                                  float4* __restrict__ out, int n4) {
    int i = blockIdx.x * blockDim.x + threadIdx.x;
    int stride = gridDim.x * blockDim.x;
    for (; i < n4; i += stride) {
        float4 a = agg[i], r = res[i], o;
        o.x = fminf(fmaxf(ALPHA_F * a.x + r.x, 0.0f), 1.0f);
        o.y = fminf(fmaxf(ALPHA_F * a.y + r.y, 0.0f), 1.0f);
        o.z = fminf(fmaxf(ALPHA_F * a.z + r.z, 0.0f), 1.0f);
        o.w = fminf(fmaxf(ALPHA_F * a.w + r.w, 0.0f), 1.0f);
        out[i] = o;
    }
}

// ---------------- all-zero-row mask + count ----------------
__global__ void lp_finalize_kernel(const float* __restrict__ out,
                                   float* __restrict__ count, float* __restrict__ mask) {
    int i = blockIdx.x * blockDim.x + threadIdx.x;
    int stride = gridDim.x * blockDim.x;
    for (; i < NN; i += stride) {
        const float4* p = reinterpret_cast<const float4*>(out + (size_t)i * DD);
        bool allz = true;
#pragma unroll
        for (int j = 0; j < DD / 4; ++j) {
            float4 v = p[j];
            allz = allz && (v.x == 0.0f) && (v.y == 0.0f) && (v.z == 0.0f) && (v.w == 0.0f);
        }
        mask[i] = allz ? 1.0f : 0.0f;
        if (allz) atomicAdd(count, 1.0f);
    }
}

extern "C" void kernel_launch(void* const* d_in, const int* in_sizes, int n_in,
                              void* d_out, int out_size, void* d_ws, size_t ws_size,
                              hipStream_t stream) {
    (void)in_sizes; (void)n_in; (void)out_size; (void)ws_size;

    const int*   train_mask = (const int*)d_in[0];
    const float* protos     = (const float*)d_in[1];
    const int*   labels     = (const int*)d_in[2];
    const int*   edges      = (const int*)d_in[3];
    const int*   row        = edges;        // edge_index[0]
    const int*   col        = edges + EE;   // edge_index[1]

    float* out   = (float*)d_out;               // labels [N*D]
    float* count = out + (size_t)NN * DD;       // num_not_updated [1]
    float* mask  = count + 1;                   // not_updated_mask [N]

    // workspace layout (floats): deg[N] | norm[E] | res[N*D] | agg[N*D]  (~58 MB)
    float* ws   = (float*)d_ws;
    float* deg  = ws;
    float* norm = ws + NN;
    float* res  = ws + NN + EE;
    float* agg  = ws + NN + EE + (size_t)NN * DD;

    const int B = 256;
    const int ND4 = NN * DD / 4;

    // degree + symmetric normalization
    lp_zero_kernel  <<<(NN + B - 1) / B, B, 0, stream>>>(deg, NN);
    lp_degree_kernel<<<(EE + B - 1) / B, B, 0, stream>>>(col, deg, EE);
    lp_dinv_kernel  <<<(NN + B - 1) / B, B, 0, stream>>>(deg, NN);
    lp_norm_kernel  <<<(EE + B - 1) / B, B, 0, stream>>>(row, col, deg, norm, EE);

    // WMMA one-hot GEMM init: 6250 waves, 8 waves per 256-thread block
    {
        int ngroups = NN / 16;                       // 6250
        int blocks  = (ngroups + 7) / 8;             // 782
        lp_init_wmma_kernel<<<blocks, 256, 0, stream>>>(train_mask, protos, labels, out, res);
    }

    // 3 propagation layers (working set ~83 MB -> fully L2-resident on 192 MB L2)
    for (int l = 0; l < NLAYERS; ++l) {
        lp_zero4_kernel  <<<(ND4 + B - 1) / B, B, 0, stream>>>((float4*)agg, ND4);
        lp_scatter_kernel<<<(EE * 16 + B - 1) / B, B, 0, stream>>>(row, col, norm, out, agg);
        lp_combine_kernel<<<(ND4 + B - 1) / B, B, 0, stream>>>((const float4*)agg,
                                                               (const float4*)res,
                                                               (float4*)out, ND4);
    }

    // mask + count
    lp_zero_kernel    <<<1, 32, 0, stream>>>(count, 1);
    lp_finalize_kernel<<<(NN + B - 1) / B, B, 0, stream>>>(out, count, mask);
}
